// Attention_70806830842616
// MI455X (gfx1250) — matrix operand order repaired and checked by reading, not compiled, not run on hardware
//
#include <hip/hip_runtime.h>

// ---------------------------------------------------------------------------
// Slot-attention for MI455X (gfx1250, wave32, WMMA).
// All six GEMMs run through one batched bf16 WMMA kernel
// (v_wmma_f32_16x16x32_bf16, f32 accumulate) with double-buffered LDS tiles
// filled by GLOBAL_LOAD_ASYNC_TO_LDS_B128 (ASYNCcnt / s_wait_asynccnt) when
// the toolchain exposes the builtin, falling back to register-staged copies.
// The 64x1024x1024 attention matrix is kept in bf16 to halve HBM traffic
// (~0.6 GB of S traffic vs ~34 GFLOP of math -> memory bound at 23.3 TB/s).
// SCALE is folded into the Q-projection epilogue; the slot-softmax row
// renormalization 1/(rowsum+eps) is folded into the A@V epilogue (v_rcp_f32).
// ---------------------------------------------------------------------------

typedef __bf16 bf16;
typedef __bf16 bf16x4 __attribute__((ext_vector_type(4)));
typedef __bf16 bf16x8 __attribute__((ext_vector_type(8)));
typedef __bf16 v16bf  __attribute__((ext_vector_type(16)));
typedef float  v8f    __attribute__((ext_vector_type(8)));
typedef float  f32x4  __attribute__((ext_vector_type(4)));
typedef int    v4i_vs __attribute__((vector_size(16)));   // builtin's pointee

#define N_HEADS   8
#define DIMH      64
#define SEQ       1024
#define BATCH     8
#define DMODEL    512
#define INNER     512
#define ATTN_SCALE 0.125f   /* 64^-0.5 */
#define EPS_SLOT  1e-7f

#define LDS_PAD   40        /* 32 bf16 + 8 pad -> 80B row stride, bank-spread */

#if defined(__gfx1250__) && __has_builtin(__builtin_amdgcn_global_load_async_to_lds_b128)
#define HAVE_ASYNC_LDS 1
#else
#define HAVE_ASYNC_LDS 0
#endif

// 16-byte global -> LDS copy (async DMA on gfx1250 when available).
__device__ __forceinline__ void cp16_g2l(bf16* l, const bf16* g) {
#if HAVE_ASYNC_LDS
  __builtin_amdgcn_global_load_async_to_lds_b128(
      (__attribute__((address_space(1))) v4i_vs*)(g),
      (__attribute__((address_space(3))) v4i_vs*)(l), 0, 0);
#else
  __builtin_prefetch(g + 64, 0, 0);
  *(bf16x8*)l = *(const bf16x8*)g;
#endif
}

__device__ __forceinline__ void wait_stage() {
#if HAVE_ASYNC_LDS && __has_builtin(__builtin_amdgcn_s_wait_asynccnt)
  __builtin_amdgcn_s_wait_asynccnt(0);
#endif
}

// ---------------------------------------------------------------------------
// f32 -> bf16 conversion (vectorized x4)
// ---------------------------------------------------------------------------
__global__ void cvt_f32_bf16(const float* __restrict__ src,
                             bf16* __restrict__ dst, int n4) {
  int i = blockIdx.x * blockDim.x + threadIdx.x;
  if (i < n4) {
    f32x4 f = ((const f32x4*)src)[i];
    bf16x4 o;
    o.x = (bf16)f.x; o.y = (bf16)f.y; o.z = (bf16)f.z; o.w = (bf16)f.w;
    ((bf16x4*)dst)[i] = o;
  }
}

// ---------------------------------------------------------------------------
// Generic batched bf16 WMMA GEMM:  C[M,N] = alpha * A[M,K] @ B[N,K]^T
// A row-major (lda == K), B row-major (ldb == K)  -> both K-contiguous,
// which matches every GEMM in this model.
// Block: 256 threads = 8 waves (wave32). Block tile 128x64, wave tile 32x32.
// Double-buffered LDS staging: A-tile 128x32, B-tile 64x32 per K-step.
// Epilogue modes:
//   0: bf16, head-split [b,h,n,dh]            (Q with alpha=SCALE, K)
//   1: bf16, head-split transposed [b,h,dh,n] (V^T so A@V is K-contiguous)
//   2: bf16, plain row-major, ld=ldOut, batch stride sO   (scores S)
//   3: bf16, head-merge [b,n,h*dh], row-scaled by 1/(rowsum+eps)  (A@V)
//   4: f32 + bias, row-major [M, 512]                      (final out proj)
// ---------------------------------------------------------------------------
__global__ __launch_bounds__(256)
void gemm_wmma_bf16(const bf16* __restrict__ A, const bf16* __restrict__ B,
                    void* __restrict__ Out, const float* __restrict__ extra,
                    int K, long long sA, long long sB, long long sO,
                    int ldOut, int mode, float alpha) {
  __shared__ bf16 ldsA[2][128 * LDS_PAD];   // 2 x 10 KB
  __shared__ bf16 ldsB[2][64 * LDS_PAD];    // 2 x  5 KB

  const int tid  = threadIdx.x;
  const int lane = tid & 31;
  const int wid  = tid >> 5;
  const int wrow = wid & 3;          // 0..3
  const int wcol = wid >> 2;         // 0..1
  const int z    = blockIdx.z;

  const bf16* Ab = A + (long long)z * sA;
  const bf16* Bb = B + (long long)z * sB;

  const int blkRow0 = blockIdx.x * 128;
  const int blkCol0 = blockIdx.y * 64;
  const int m16 = lane & 15;
  const int hi  = lane >> 4;         // lane-half select for K layout

  // Stage one K-step tile pair into LDS buffer `buf` (async on gfx1250).
  // A-tile: 128 rows x 32 cols = 512 x 16B chunks (2/thread).
  // B-tile:  64 rows x 32 cols = 256 x 16B chunks (1/thread).
  auto stage = [&](int buf, int kc) {
#pragma unroll
    for (int c = tid; c < 512; c += 256) {
      int r = c >> 2, part = c & 3;
      cp16_g2l(&ldsA[buf][r * LDS_PAD + part * 8],
               Ab + (long long)(blkRow0 + r) * K + kc + part * 8);
    }
    {
      int r = tid >> 2, part = tid & 3;
      cp16_g2l(&ldsB[buf][r * LDS_PAD + part * 8],
               Bb + (long long)(blkCol0 + r) * K + kc + part * 8);
    }
  };

  v8f acc[2][2] = {};
  const int nk = K >> 5;
  stage(0, 0);
  int cur = 0;

  for (int s = 0; s < nk; ++s) {
    wait_stage();          // this wave's async copies into `cur` complete
    __syncthreads();       // ...and every other wave's too
    if (s + 1 < nk) stage(cur ^ 1, (s + 1) << 5);

    // Fragments from LDS, per ISA 16-bit A/B VGPR layouts.
    const bf16* a0 = &ldsA[cur][(wrow * 32 + m16) * LDS_PAD + hi * 8];
    const bf16* b0 = &ldsB[cur][(wcol * 32 + m16) * LDS_PAD + hi * 8];
    v16bf afrag[2], bfrag[2];
#pragma unroll
    for (int t = 0; t < 2; ++t) {
      const bf16* ap = a0 + t * 16 * LDS_PAD;
      afrag[t] = __builtin_shufflevector(*(const bf16x8*)ap,
                                         *(const bf16x8*)(ap + 16),
                   0,1,2,3,4,5,6,7,8,9,10,11,12,13,14,15);
      const bf16* bp = b0 + t * 16 * LDS_PAD;
      bfrag[t] = __builtin_shufflevector(*(const bf16x8*)bp,
                                         *(const bf16x8*)(bp + 16),
                   0,1,2,3,4,5,6,7,8,9,10,11,12,13,14,15);
    }
#pragma unroll
    for (int i = 0; i < 2; ++i)
#pragma unroll
      for (int j = 0; j < 2; ++j)
        acc[i][j] = __builtin_amdgcn_wmma_f32_16x16x32_bf16(
            false, afrag[i], false, bfrag[j],
            (short)0, acc[i][j], false, false);

    __syncthreads();       // everyone done reading `cur` before it's reused
    cur ^= 1;
  }

  // Epilogue. C layout: VGPR r, lanes 0-15 -> (M=r, N=lane);
  // lanes 16-31 -> (M=8+r, N=lane-16).
#pragma unroll
  for (int i = 0; i < 2; ++i)
#pragma unroll
    for (int j = 0; j < 2; ++j)
#pragma unroll
      for (int r = 0; r < 8; ++r) {
        int grow = blkRow0 + wrow * 32 + i * 16 + hi * 8 + r;
        int gcol = blkCol0 + wcol * 32 + j * 16 + m16;
        float v = acc[i][j][r] * alpha;
        if (mode == 0) {            // [b,h,n,dh]
          int b = grow >> 10, ii = grow & 1023;
          int h = gcol >> 6,  d  = gcol & 63;
          ((bf16*)Out)[(((long long)(b * N_HEADS + h) * SEQ + ii) * DIMH) + d]
              = (bf16)v;
        } else if (mode == 1) {     // [b,h,dh,n]
          int b = grow >> 10, ii = grow & 1023;
          int h = gcol >> 6,  d  = gcol & 63;
          ((bf16*)Out)[(((long long)(b * N_HEADS + h) * DIMH + d) * SEQ) + ii]
              = (bf16)v;
        } else if (mode == 2) {     // plain bf16 (scores S)
          ((bf16*)Out)[sO * z + (long long)grow * ldOut + gcol] = (bf16)v;
        } else if (mode == 3) {     // head-merge + row renormalization
          int b = z >> 3, h = z & 7;
          float rs = extra[(long long)z * SEQ + grow];
          float o = v * __builtin_amdgcn_rcpf(rs + EPS_SLOT);
          ((bf16*)Out)[((long long)(b * SEQ + grow) * INNER) + h * DIMH + gcol]
              = (bf16)o;
        } else {                    // mode 4: f32 + bias (final projection)
          ((float*)Out)[(long long)grow * DMODEL + gcol] = v + extra[gcol];
        }
      }
}

// ---------------------------------------------------------------------------
// Column stats for slot softmax (softmax over the QUERY axis):
// one thread per column j of S[bh] -> online max + sum(exp) over i=0..1023.
// Threads in a block scan adjacent columns -> coalesced strided reads.
// ---------------------------------------------------------------------------
__global__ __launch_bounds__(256)
void colstats(const bf16* __restrict__ S, float* __restrict__ cmax,
              float* __restrict__ csum) {
  int col = blockIdx.x * blockDim.x + threadIdx.x;   // 0..65535
  int bh = col >> 10, j = col & 1023;
  const bf16* p = S + (long long)bh * SEQ * SEQ + j;
  float mx = -3.4e38f, sm = 0.f;
  for (int i = 0; i < SEQ; ++i) {
    float s = (float)p[(long long)i * SEQ];
    if (s > mx) { sm = sm * __expf(mx - s); mx = s; }
    sm += __expf(s - mx);
  }
  cmax[col] = mx;
  csum[col] = sm;    // >= 1 always (max term contributes exp(0)=1)
}

// ---------------------------------------------------------------------------
// Row pass: A'_ij = exp(S_ij - colmax_j)/colsum_j written back into S (bf16),
// and rowsum_i = sum_j A'_ij via wave32 shuffle reduction (one wave per row).
// ---------------------------------------------------------------------------
__global__ __launch_bounds__(256)
void rownorm(bf16* __restrict__ S, const float* __restrict__ cmax,
             const float* __restrict__ csum, float* __restrict__ rsum) {
  int lane = threadIdx.x & 31;
  int wrow = blockIdx.x * 8 + (threadIdx.x >> 5);    // global row 0..65535
  int bh = wrow >> 10, i = wrow & 1023;
  bf16* p = S + (long long)bh * SEQ * SEQ + (long long)i * SEQ;
  const float* cm = cmax + bh * SEQ;
  const float* cs = csum + bh * SEQ;
  float acc = 0.f;
  for (int j = lane; j < SEQ; j += 32) {
    float a = __expf((float)p[j] - cm[j]) * __builtin_amdgcn_rcpf(cs[j]);
    p[j] = (bf16)a;
    acc += a;
  }
#pragma unroll
  for (int off = 16; off; off >>= 1) acc += __shfl_xor(acc, off, 32);
  if (lane == 0) rsum[wrow] = acc;
}

// ---------------------------------------------------------------------------
// Host-side orchestration
// ---------------------------------------------------------------------------
extern "C" void kernel_launch(void* const* d_in, const int* in_sizes, int n_in,
                              void* d_out, int out_size, void* d_ws, size_t ws_size,
                              hipStream_t stream) {
  const float* x   = (const float*)d_in[0];   // [8,1024,512]
  const float* ctx = (const float*)d_in[1];   // [8,1024,512]
  const float* Wq  = (const float*)d_in[2];   // [512,512]
  const float* Wk  = (const float*)d_in[3];
  const float* Wv  = (const float*)d_in[4];
  const float* Wo  = (const float*)d_in[5];
  const float* bo  = (const float*)d_in[6];   // [512]
  float* out = (float*)d_out;                 // [8,1024,512]

  const long long NX  = (long long)BATCH * SEQ * DMODEL;  // 4,194,304
  const long long NW  = (long long)INNER * DMODEL;        //   262,144
  const long long NQ  = (long long)BATCH * N_HEADS * SEQ * DIMH;
  const long long NS  = (long long)BATCH * N_HEADS * SEQ * SEQ;
  const long long NST = (long long)BATCH * N_HEADS * SEQ;

  char* w = (char*)d_ws;
  bf16* Xb  = (bf16*)w;  w += NX * 2;       // bf16 x
  bf16* Cb  = (bf16*)w;  w += NX * 2;       // bf16 context
  bf16* Wqb = (bf16*)w;  w += NW * 2;
  bf16* Wkb = (bf16*)w;  w += NW * 2;
  bf16* Wvb = (bf16*)w;  w += NW * 2;
  bf16* Wob = (bf16*)w;  w += NW * 2;
  bf16* Qw  = (bf16*)w;  w += NQ * 2;       // [b,h,n,dh], pre-scaled
  bf16* Kw  = (bf16*)w;  w += NQ * 2;       // [b,h,m,dh]
  bf16* Vt  = (bf16*)w;  w += NQ * 2;       // [b,h,dh,m]
  bf16* Ob  = (bf16*)w;  w += NX * 2;       // [b,n,inner] head-merged
  float* cmax = (float*)w; w += NST * 4;
  float* csum = (float*)w; w += NST * 4;
  float* rsum = (float*)w; w += NST * 4;
  bf16* Sb  = (bf16*)w;  w += NS * 2;       // [bh,n,m] attention matrix

  auto cvt = [&](const float* s, bf16* d, long long n) {
    int n4 = (int)(n / 4);
    cvt_f32_bf16<<<(n4 + 255) / 256, 256, 0, stream>>>(s, d, n4);
  };
  cvt(x,   Xb,  NX);
  cvt(ctx, Cb,  NX);
  cvt(Wq,  Wqb, NW);
  cvt(Wk,  Wkb, NW);
  cvt(Wv,  Wvb, NW);
  cvt(Wo,  Wob, NW);

  // Projections: [8192,512] @ [512,512]^T ; grid = (8192/128, 512/64)
  gemm_wmma_bf16<<<dim3(64, 8, 1), 256, 0, stream>>>(
      Xb, Wqb, Qw, nullptr, DMODEL, 0, 0, 0, 0, /*mode*/0, ATTN_SCALE);
  gemm_wmma_bf16<<<dim3(64, 8, 1), 256, 0, stream>>>(
      Cb, Wkb, Kw, nullptr, DMODEL, 0, 0, 0, 0, /*mode*/0, 1.0f);
  gemm_wmma_bf16<<<dim3(64, 8, 1), 256, 0, stream>>>(
      Cb, Wvb, Vt, nullptr, DMODEL, 0, 0, 0, 0, /*mode*/1, 1.0f);

  // Scores: per (b,h)  S = Q @ K^T  (K=64); grid = (1024/128, 1024/64, 64)
  gemm_wmma_bf16<<<dim3(8, 16, 64), 256, 0, stream>>>(
      Qw, Kw, Sb, nullptr, DIMH,
      (long long)SEQ * DIMH, (long long)SEQ * DIMH, (long long)SEQ * SEQ,
      SEQ, /*mode*/2, 1.0f);

  // Slot softmax: column stats then row pass (writes exp'd probs + rowsums).
  colstats<<<dim3(256, 1, 1), 256, 0, stream>>>(Sb, cmax, csum);
  rownorm<<<dim3(8192, 1, 1), 256, 0, stream>>>(Sb, cmax, csum, rsum);

  // A @ V with fused 1/(rowsum+eps): per (b,h) [1024,1024] @ [64,1024]^T
  gemm_wmma_bf16<<<dim3(8, 1, 64), 256, 0, stream>>>(
      Sb, Vt, Ob, rsum, SEQ,
      (long long)SEQ * SEQ, (long long)DIMH * SEQ, 0,
      0, /*mode*/3, 1.0f);

  // Output projection: [8192,512] @ [512,512]^T + bias -> f32 d_out
  gemm_wmma_bf16<<<dim3(64, 8, 1), 256, 0, stream>>>(
      Ob, Wob, (void*)out, bo, INNER, 0, 0, 0, 0, /*mode*/4, 1.0f);
}
